// PhysicsAttentionStructuredMesh2D_6356551598267
// MI455X (gfx1250) — matrix operand order repaired and checked by reading, not compiled, not run on hardware
//
#include <hip/hip_runtime.h>
#include <hip/hip_bf16.h>
#include <stdint.h>

typedef __bf16 bf16;
typedef __attribute__((ext_vector_type(16))) __bf16 v16bf;
typedef __attribute__((ext_vector_type(8)))  float  v8f;

constexpr int kB     = 8;
constexpr int kHm    = 128;
constexpr int kWm    = 128;
constexpr int kN     = 16384;   // Hm*Wm
constexpr int kDim   = 128;
constexpr int kHeads = 8;
constexpr int kDh    = 64;
constexpr int kInner = 512;     // heads*dh
constexpr int kG     = 64;      // slice_num
constexpr int kCout2 = 1024;    // both convs fused: [0,512)=fx, [512,1024)=x
constexpr int kKred  = 1152;    // 9 * 128 im2col depth
constexpr int kSeg   = 8;       // K-split for slice_token reduction

// ---------------- WMMA fragment helpers (bf16 16x16x32, wave32) ----------------

static __device__ __forceinline__ v16bf frag_from(uint4 lo, uint4 hi) {
  union { v16bf v; uint4 q[2]; } u;
  u.q[0] = lo; u.q[1] = hi;
  return u.v;
}

// A-matrix 16x32 bf16: lane%16 = row M; grp = lane/16 selects K offset (0 or 8).
// VGPR0..3 = K{g8..g8+7}, VGPR4..7 = K{16+g8..16+g8+7}; two contiguous 16B loads.
static __device__ __forceinline__ v16bf load_frag_a(const bf16* rowp, int grp) {
  uint4 lo = *(const uint4*)(rowp + grp * 8);
  uint4 hi = *(const uint4*)(rowp + 16 + grp * 8);
  return frag_from(lo, hi);
}

static __device__ __forceinline__ v16bf zero_frag() {
  uint4 z = {0u, 0u, 0u, 0u};
  return frag_from(z, z);
}

// B-matrix 32x16 bf16: lane%16 = col N; grp = lane/16; elements = K{g8..g8+7}, K{16+g8..16+g8+7}
// (rowp points at row kbase+lane of a row-major (K, ld) matrix).
static __device__ __forceinline__ v16bf load_frag_b(const bf16* rowp, int ld) {
  const int lane = threadIdx.x & 31;
  const int g = lane >> 4;
  const bf16* p = rowp - (size_t)lane * ld + (lane & 15);
  v16bf v;
  for (int s = 0; s < 8; ++s) {
    v[s] = p[(size_t)(g * 8 + s) * ld];
    v[s + 8] = p[(size_t)(16 + g * 8 + s) * ld];
  }
  return v;
}

static __device__ __forceinline__ v8f wmma_bf16(v16bf a, v16bf b, v8f c) {
  return __builtin_amdgcn_wmma_f32_16x16x32_bf16(false, a, false, b, (short)0, c,
                                                 false, false);
}

// ---------------- pack / convert kernels ----------------

__global__ void pack_x_bf16(const float* __restrict__ x, bf16* __restrict__ xb, int n) {
  int i = blockIdx.x * blockDim.x + threadIdx.x;
  if (i < n) xb[i] = (bf16)x[i];
}

// Fuse conv_fx_w and conv_x_w into one (1152 x 1024) row-major bf16 B matrix.
// (K,K,Cin,Cout) flat is already (kk*128+ci, co) row-major.
__global__ void pack_wconv(const float* __restrict__ fxw, const float* __restrict__ xw,
                           bf16* __restrict__ wcomb) {
  int i = blockIdx.x * blockDim.x + threadIdx.x;
  if (i >= kKred * kCout2) return;
  int r = i >> 10, c = i & 1023;
  float v = (c < kInner) ? fxw[r * kInner + c] : xw[r * kInner + (c - kInner)];
  wcomb[i] = (bf16)v;
}

// slice_w (G, DH) -> swt_w (DH, G) bf16  (B operand of logits GEMM)
__global__ void pack_slicew(const float* __restrict__ sw, bf16* __restrict__ swt) {
  int i = blockIdx.x * blockDim.x + threadIdx.x;
  if (i >= kDh * kG) return;
  int d = i >> 6, g = i & 63;
  swt[i] = (bf16)sw[g * kDh + d];
}

// out_w (DIM, INNER) -> wT (INNER, DIM) bf16 (B operand of final GEMM)
__global__ void pack_outw(const float* __restrict__ ow, bf16* __restrict__ wt) {
  int i = blockIdx.x * blockDim.x + threadIdx.x;
  if (i >= kInner * kDim) return;
  int k = i >> 7, c = i & 127;
  wt[i] = (bf16)ow[c * kInner + k];
}

// ---------------- conv (implicit GEMM, fused fx+x convs) ----------------
// M = B*N pixels, Ncols = 1024, K = 9*128. One wave = 32x64 register tile:
// 2 A frags + 4 B frags -> 8 WMMAs per K-chunk.
// A 32-pixel tile is one contiguous run inside a single mesh row (32 | 128),
// so batch b, row y and the y-bounds test are wave-uniform (SGPR-resident).

__global__ __launch_bounds__(256, 2)
void conv_wmma(const bf16* __restrict__ xb, const bf16* __restrict__ wcomb,
               const float* __restrict__ bfx, const float* __restrict__ bx,
               bf16* __restrict__ mid) {
  const int lane = threadIdx.x & 31;
  const int wid  = blockIdx.x * (blockDim.x >> 5) + (threadIdx.x >> 5);
  const int nt   = wid & 15;      // 16 cout tiles of 64
  const int mt   = wid >> 4;      // 4096 pixel tiles of 32
  const int n0   = nt * 64;
  const int grp  = lane >> 4;
  const int m0   = mt * 32;              // first pixel of tile
  const int b    = m0 >> 14;             // uniform
  const int npix = m0 & (kN - 1);
  const int y    = npix >> 7;            // uniform
  const int x0   = npix & 127;           // uniform
  const int xl   = x0 + (lane & 15);     // per-lane x of sub-tile 0

  v8f acc[2][4] = {};
  for (int tap = 0; tap < 9; ++tap) {
    const int dy = tap / 3 - 1, dx = tap % 3 - 1;
    const int yy = y + dy;
    const bool vy = (yy >= 0) & (yy < kHm);          // uniform
    const bf16* rowbase = xb + (size_t)(b * kHm + yy) * kWm * kDim;
    const int xx0 = xl + dx;
    const int xx1 = xx0 + 16;
    const bool v0 = vy & (xx0 >= 0) & (xx0 < kWm);
    const bool v1 = vy & (xx1 >= 0) & (xx1 < kWm);
    const bf16* arow0 = rowbase + (size_t)xx0 * kDim;
    const bf16* arow1 = rowbase + (size_t)xx1 * kDim;
    for (int c = 0; c < 4; ++c) {
      const int kc = tap * 4 + c;
      v16bf a0 = v0 ? load_frag_a(arow0 + c * 32, grp) : zero_frag();
      v16bf a1 = v1 ? load_frag_a(arow1 + c * 32, grp) : zero_frag();
      const bf16* brow = wcomb + (size_t)(kc * 32 + lane) * kCout2 + n0;
      __builtin_prefetch(brow + 32 * kCout2, 0, 1);   // next K-chunk of weights
      for (int f = 0; f < 4; ++f) {
        v16bf bfr = load_frag_b(brow + f * 16, kCout2);
        acc[0][f] = wmma_bf16(a0, bfr, acc[0][f]);
        acc[1][f] = wmma_bf16(a1, bfr, acc[1][f]);
      }
    }
  }
  for (int f = 0; f < 4; ++f) {
    const int col = n0 + f * 16 + (lane & 15);
    const float bias = (col < kInner) ? bfx[col] : bx[col - kInner];
    for (int mi = 0; mi < 2; ++mi)
      for (int r = 0; r < 8; ++r) {
        const int nn = npix + mi * 16 + r + (grp << 3);  // same mesh row
        mid[(size_t)(b * kN + nn) * kCout2 + col] = (bf16)(acc[mi][f][r] + bias);
      }
  }
}

// ---------------- logits GEMM + fused softmax over G ----------------
// Per (b,h): logits(N x 64) = x_mid(N x 64) * swt_w(64 x 64); softmax rows;
// writes sw (b,h,N,G) and transposed swT (b,h,G,N), both bf16.

__global__ __launch_bounds__(256, 4)
void logits_softmax(const bf16* __restrict__ mid,
                    const bf16* __restrict__ swt_w,
                    const float* __restrict__ slice_b,
                    const float* __restrict__ temperature,
                    bf16* __restrict__ sw, bf16* __restrict__ swT) {
  const int lane = threadIdx.x & 31;
  const int wid  = blockIdx.x * (blockDim.x >> 5) + (threadIdx.x >> 5);
  const int mtl  = wid & 1023;     // n-tile within (b,h)
  const int bh   = wid >> 10;      // 0..63
  const int b    = bh >> 3, h = bh & 7;
  const int grp  = lane >> 4;
  const int nrow = mtl * 16 + (lane & 15);
  const bf16* arow = mid + (size_t)(b * kN + nrow) * kCout2 + kInner + h * kDh;

  v8f acc[4] = {{}, {}, {}, {}};
  for (int kc = 0; kc < 2; ++kc) {
    v16bf af = load_frag_a(arow + kc * 32, grp);
    for (int f = 0; f < 4; ++f) {
      const bf16* brow = swt_w + (size_t)(kc * 32 + lane) * kG + f * 16;
      acc[f] = wmma_bf16(af, load_frag_b(brow, kG), acc[f]);
    }
  }
  float t = fminf(fmaxf(temperature[h], 0.1f), 5.0f);
  const float invt = 1.0f / t;

  float xv[4][8];
  for (int f = 0; f < 4; ++f) {
    const float sb = slice_b[f * 16 + (lane & 15)];
    for (int r = 0; r < 8; ++r) xv[f][r] = (acc[f][r] + sb) * invt;
  }
  // Row softmax over 64 cols. Shfl masks 1..8 stay inside each 16-lane half,
  // so lanes 0-15 reduce row r while lanes 16-31 reduce row r+8 in parallel.
  for (int r = 0; r < 8; ++r) {
    float mx = xv[0][r];
    for (int f = 1; f < 4; ++f) mx = fmaxf(mx, xv[f][r]);
    for (int off = 1; off < 16; off <<= 1) mx = fmaxf(mx, __shfl_xor(mx, off, 32));
    float s = 0.f;
    for (int f = 0; f < 4; ++f) { xv[f][r] = __expf(xv[f][r] - mx); s += xv[f][r]; }
    for (int off = 1; off < 16; off <<= 1) s += __shfl_xor(s, off, 32);
    const float inv = 1.0f / s;
    for (int f = 0; f < 4; ++f) xv[f][r] *= inv;
  }
  const size_t swb  = (size_t)bh * kN * kG;
  const size_t swTb = (size_t)bh * kG * kN;
  for (int f = 0; f < 4; ++f) {
    const int g = f * 16 + (lane & 15);
    for (int r = 0; r < 8; ++r) {
      const int nn = mtl * 16 + r + (grp << 3);
      const bf16 w = (bf16)xv[f][r];
      sw[swb + (size_t)nn * kG + g]  = w;
      swT[swTb + (size_t)g * kN + nn] = w;
    }
  }
}

// slice_norm[bh][g] = sum_n sw[bh][n][g]  (deterministic fixed-order reduction)
__global__ void slice_norm_reduce(const bf16* __restrict__ swT, float* __restrict__ snorm) {
  const int lane = threadIdx.x & 31;
  const int wid  = blockIdx.x * (blockDim.x >> 5) + (threadIdx.x >> 5); // bh*64+g
  const bf16* row = swT + (size_t)wid * kN;
  float s = 0.f;
  for (int n = lane; n < kN; n += 32) s += (float)row[n];
  for (int off = 16; off; off >>= 1) s += __shfl_xor(s, off, 32);
  if (lane == 0) snorm[wid] = s;
}

// ------ slice_token = swT(G x N) * fx_mid(N x DH); K split into kSeg segments ---
// One wave = 16 G-rows x full DH (4 accumulators, A frag reused 4x).

__global__ __launch_bounds__(256, 4)
void slice_token_partial(const bf16* __restrict__ swT,
                         const bf16* __restrict__ mid,
                         float* __restrict__ part) {
  const int lane = threadIdx.x & 31;
  const int wid  = blockIdx.x * (blockDim.x >> 5) + (threadIdx.x >> 5);
  const int tg   = wid & 3;
  const int bh   = (wid >> 2) & 63;
  const int seg  = wid >> 8;                   // 0..kSeg-1
  const int b    = bh >> 3, h = bh & 7;
  const int grp  = lane >> 4;
  const int nseg = kN / kSeg;                  // 2048 rows per segment
  const bf16* arow = swT + ((size_t)bh * kG + tg * 16 + (lane & 15)) * kN + seg * nseg;

  v8f acc[4] = {{}, {}, {}, {}};
  for (int kc = 0; kc < nseg / 32; ++kc) {     // 64 chunks
    v16bf af = load_frag_a(arow + kc * 32, grp);
    const int n = seg * nseg + kc * 32 + lane;
    const bf16* brow = mid + (size_t)(b * kN + n) * kCout2 + h * kDh;
    for (int f = 0; f < 4; ++f)
      acc[f] = wmma_bf16(af, load_frag_b(brow + f * 16, kCout2), acc[f]);
  }
  float* dst = part + (size_t)seg * kB * kHeads * kG * kDh + (size_t)bh * kG * kDh;
  for (int f = 0; f < 4; ++f) {
    const int c = f * 16 + (lane & 15);
    for (int r = 0; r < 8; ++r) {
      const int g = tg * 16 + r + (grp << 3);
      dst[(size_t)g * kDh + c] = acc[f][r];
    }
  }
}

__global__ void slice_token_finalize(const float* __restrict__ part,
                                     const float* __restrict__ snorm,
                                     float* __restrict__ stok) {
  const int i = blockIdx.x * blockDim.x + threadIdx.x;
  if (i >= kB * kHeads * kG * kDh) return;
  float s = 0.f;
  for (int seg = 0; seg < kSeg; ++seg)
    s += part[(size_t)seg * kB * kHeads * kG * kDh + i];
  const int g  = (i >> 6) & 63;   // i = ((bh*64+g)*64 + c)
  const int bh = i >> 12;
  stok[i] = s / (snorm[bh * kG + g] + 1e-5f);
}

// ---------------- tiny 64x64 q/k/v + attention, per (b,h) block ----------------

__global__ void slice_attention(const float* __restrict__ stok,
                                const float* __restrict__ wq, const float* __restrict__ wk,
                                const float* __restrict__ wv, bf16* __restrict__ oslice) {
  __shared__ float sST[kG * kDh];
  __shared__ float sQ[kG * kDh];
  __shared__ float sK[kG * kDh];
  __shared__ float sV[kG * kDh];
  const int bh = blockIdx.x;
  const int tid = threadIdx.x;
  for (int i = tid; i < kG * kDh; i += blockDim.x)
    sST[i] = stok[(size_t)bh * kG * kDh + i];
  __syncthreads();
  for (int i = tid; i < kG * kDh; i += blockDim.x) {
    const int g = i >> 6, d = i & 63;
    float aq = 0.f, ak = 0.f, av = 0.f;
    for (int c = 0; c < kDh; ++c) {
      const float s = sST[g * kDh + c];
      aq += s * wq[d * kDh + c];
      ak += s * wk[d * kDh + c];
      av += s * wv[d * kDh + c];
    }
    sQ[i] = aq; sK[i] = ak; sV[i] = av;
  }
  __syncthreads();
  if (tid < kG) {
    const float scale = 0.125f;  // 1/sqrt(64)
    float s[kG];
    float mx = -1e30f;
    for (int j = 0; j < kG; ++j) {
      float d0 = 0.f;
      for (int c = 0; c < kDh; ++c) d0 += sQ[tid * kDh + c] * sK[j * kDh + c];
      s[j] = d0 * scale;
      mx = fmaxf(mx, s[j]);
    }
    float sum = 0.f;
    for (int j = 0; j < kG; ++j) { s[j] = __expf(s[j] - mx); sum += s[j]; }
    const float inv = 1.0f / sum;
    for (int d = 0; d < kDh; ++d) {
      float o = 0.f;
      for (int j = 0; j < kG; ++j) o += s[j] * sV[j * kDh + d];
      oslice[(size_t)bh * kG * kDh + tid * kDh + d] = (bf16)(o * inv);
    }
  }
}

// -------- out_x = sw(N x G) * out_slice(G x DH), per (b,h); 16x64 tile ---------

__global__ __launch_bounds__(256, 4)
void out_x_wmma(const bf16* __restrict__ sw, const bf16* __restrict__ oslice,
                bf16* __restrict__ outx) {
  const int lane = threadIdx.x & 31;
  const int wid  = blockIdx.x * (blockDim.x >> 5) + (threadIdx.x >> 5);
  const int mtl  = wid & 1023;              // 1024 m-tiles
  const int bh   = wid >> 10;
  const int b    = bh >> 3, h = bh & 7;
  const int grp  = lane >> 4;
  const bf16* arow = sw + ((size_t)bh * kN + mtl * 16 + (lane & 15)) * kG;

  v8f acc[4] = {{}, {}, {}, {}};
  for (int kc = 0; kc < 2; ++kc) {
    v16bf af = load_frag_a(arow + kc * 32, grp);
    const bf16* brow = oslice + (size_t)bh * kG * kDh + (size_t)(kc * 32 + lane) * kDh;
    for (int f = 0; f < 4; ++f)
      acc[f] = wmma_bf16(af, load_frag_b(brow + f * 16, kDh), acc[f]);
  }
  for (int f = 0; f < 4; ++f) {
    const int c = h * kDh + f * 16 + (lane & 15);
    for (int r = 0; r < 8; ++r) {
      const int nn = mtl * 16 + r + (grp << 3);
      outx[(size_t)(b * kN + nn) * kInner + c] = (bf16)acc[f][r];
    }
  }
}

// ------- final: out = out_x(BN x 512) * wT(512 x 128) + bias; 16x128 tile ------

__global__ __launch_bounds__(256, 2)
void final_wmma(const bf16* __restrict__ outx, const bf16* __restrict__ wT,
                const float* __restrict__ outb, float* __restrict__ out) {
  const int lane = threadIdx.x & 31;
  const int mtl  = blockIdx.x * (blockDim.x >> 5) + (threadIdx.x >> 5); // 8192 tiles
  const int grp  = lane >> 4;
  const bf16* arow = outx + ((size_t)mtl * 16 + (lane & 15)) * kInner;

  v8f acc[8] = {};
  for (int kc = 0; kc < kInner / 32; ++kc) {
    v16bf af = load_frag_a(arow + kc * 32, grp);
    const bf16* brow = wT + (size_t)(kc * 32 + lane) * kDim;
    for (int f = 0; f < 8; ++f)
      acc[f] = wmma_bf16(af, load_frag_b(brow + f * 16, kDim), acc[f]);
  }
  for (int f = 0; f < 8; ++f) {
    const int col = f * 16 + (lane & 15);
    const float bias = outb[col];
    for (int r = 0; r < 8; ++r) {
      const int m = mtl * 16 + r + (grp << 3);
      out[(size_t)m * kDim + col] = acc[f][r] + bias;
    }
  }
}

// ---------------- host launcher ----------------

extern "C" void kernel_launch(void* const* d_in, const int* in_sizes, int n_in,
                              void* d_out, int out_size, void* d_ws, size_t ws_size,
                              hipStream_t stream) {
  (void)in_sizes; (void)n_in; (void)out_size; (void)ws_size;
  const float* x    = (const float*)d_in[0];
  const float* cfxw = (const float*)d_in[1];
  const float* cfxb = (const float*)d_in[2];
  const float* cxw  = (const float*)d_in[3];
  const float* cxb  = (const float*)d_in[4];
  const float* slw  = (const float*)d_in[5];
  const float* slb  = (const float*)d_in[6];
  const float* temp = (const float*)d_in[7];
  const float* wq   = (const float*)d_in[8];
  const float* wk   = (const float*)d_in[9];
  const float* wv   = (const float*)d_in[10];
  const float* outw = (const float*)d_in[11];
  const float* outb = (const float*)d_in[12];
  float* out = (float*)d_out;

  uint8_t* ws = (uint8_t*)d_ws;
  size_t off = 0;
  auto alloc = [&](size_t bytes) -> void* {
    void* p = (void*)(ws + off);
    off += (bytes + 255) & ~(size_t)255;
    return p;
  };
  bf16* xb     = (bf16*)alloc((size_t)kB * kN * kDim * 2);
  bf16* wcomb  = (bf16*)alloc((size_t)kKred * kCout2 * 2);
  bf16* swt_w  = (bf16*)alloc((size_t)kDh * kG * 2);
  bf16* wTp    = (bf16*)alloc((size_t)kInner * kDim * 2);
  bf16* mid    = (bf16*)alloc((size_t)kB * kN * kCout2 * 2);
  bf16* sw     = (bf16*)alloc((size_t)kB * kHeads * kN * kG * 2);
  bf16* swT    = (bf16*)alloc((size_t)kB * kHeads * kG * kN * 2);
  float* snorm = (float*)alloc((size_t)kB * kHeads * kG * 4);
  float* part  = (float*)alloc((size_t)kSeg * kB * kHeads * kG * kDh * 4);
  float* stok  = (float*)alloc((size_t)kB * kHeads * kG * kDh * 4);
  bf16* osl    = (bf16*)alloc((size_t)kB * kHeads * kG * kDh * 2);
  bf16* outx   = (bf16*)alloc((size_t)kB * kN * kInner * 2);

  const int TPB = 256;
  pack_x_bf16<<<(kB * kN * kDim + TPB - 1) / TPB, TPB, 0, stream>>>(x, xb, kB * kN * kDim);
  pack_wconv<<<(kKred * kCout2 + TPB - 1) / TPB, TPB, 0, stream>>>(cfxw, cxw, wcomb);
  pack_slicew<<<(kDh * kG + TPB - 1) / TPB, TPB, 0, stream>>>(slw, swt_w);
  pack_outw<<<(kInner * kDim + TPB - 1) / TPB, TPB, 0, stream>>>(outw, wTp);

  // conv: 4096 m-tiles(32px) * 16 n-tiles(64ch) = 65536 wave-tiles, 8 waves/block
  conv_wmma<<<8192, TPB, 0, stream>>>(xb, wcomb, cfxb, cxb, mid);
  // logits+softmax: 64 bh * 1024 n-tiles = 65536 wave-tiles
  logits_softmax<<<8192, TPB, 0, stream>>>(mid, swt_w, slb, temp, sw, swT);
  // slice_norm: 4096 (bh,g) rows, one wave each
  slice_norm_reduce<<<512, TPB, 0, stream>>>(swT, snorm);
  // slice_token: 8 segs * 64 bh * 4 g-tiles = 2048 wave-tiles + finalize
  slice_token_partial<<<256, TPB, 0, stream>>>(swT, mid, part);
  slice_token_finalize<<<(kB * kHeads * kG * kDh + TPB - 1) / TPB, TPB, 0, stream>>>(
      part, snorm, stok);
  // tiny attention, one block per (b,h)
  slice_attention<<<kB * kHeads, 256, 0, stream>>>(stok, wq, wk, wv, osl);
  // out_x: 64 bh * 1024 m-tiles = 65536 wave-tiles
  out_x_wmma<<<8192, TPB, 0, stream>>>(sw, osl, outx);
  // final: 8192 row-tiles (16 x 128 each)
  final_wmma<<<1024, TPB, 0, stream>>>(outx, wTp, outb, out);
}
